// ContrastiveSparseRepresentation_64029372449367
// MI455X (gfx1250) — matrix operand order repaired and checked by reading, not compiled, not run on hardware
//
#include <hip/hip_runtime.h>

typedef __attribute__((ext_vector_type(16))) _Float16 v16h;
typedef __attribute__((ext_vector_type(8)))  _Float16 v8h;
typedef __attribute__((ext_vector_type(4)))  _Float16 v4h;
typedef __attribute__((ext_vector_type(8)))  float    v8f;
typedef __attribute__((ext_vector_type(4)))  unsigned int v4u;
typedef __attribute__((ext_vector_type(8)))  int      v8i;
typedef __attribute__((ext_vector_type(4)))  int      v4i;

static constexpr int B_ROWS = 32768;
static constexpr int D_IN   = 768;
static constexpr int D_OUT  = 4096;
static constexpr int ACTIVE = 64;
static constexpr int A_PITCH = D_IN + 8;     // non-TDM path: pad 8 halves/row
static constexpr int BISECT_ITERS = 28;

// TDM LDS layout: 16 pad bytes after every 1024 data bytes (512 halves).
__device__ __forceinline__ int tdm_off(int i) { return 2 * i + 16 * (i >> 9); }

// ---------------------------------------------------------------------------
// Kernel 1: W [768,4096] f32 row-major  ->  Wt [4096,768] f16 row-major.
// ---------------------------------------------------------------------------
__global__ __launch_bounds__(256) void wconv_kernel(const float* __restrict__ W,
                                                    _Float16* __restrict__ Wt) {
    __shared__ float tile[32][33];
    const int tx = threadIdx.x, ty = threadIdx.y;
    const int n0 = blockIdx.x * 32, k0 = blockIdx.y * 32;
#pragma unroll
    for (int i = 0; i < 4; ++i)
        tile[ty + i * 8][tx] = W[(size_t)(k0 + ty + i * 8) * D_OUT + (n0 + tx)];
    __syncthreads();
#pragma unroll
    for (int i = 0; i < 4; ++i)
        Wt[(size_t)(n0 + ty + i * 8) * D_IN + (k0 + tx)] =
            (_Float16)tile[tx][ty + i * 8];
}

// ---------------------------------------------------------------------------
// Kernel 1b: x f32 -> f16 (feeds the TDM A-tile loads).
// ---------------------------------------------------------------------------
__global__ __launch_bounds__(256) void xconv_kernel(const float* __restrict__ x,
                                                    _Float16* __restrict__ xh) {
    const size_t i = ((size_t)blockIdx.x * 256 + threadIdx.x) * 4;
    const float4 v = *(const float4*)(x + i);
    v4h h = {(_Float16)v.x, (_Float16)v.y, (_Float16)v.z, (_Float16)v.w};
    *(v4h*)(xh + i) = h;
}

// ---------------------------------------------------------------------------
// Kernel 2: fused GEMM (f16 WMMA) + LayerNorm + top-64 + L2 norm.
// USE_TDM: A tile DMA'd into LDS by the Tensor Data Mover (wave 0 issues it).
// ---------------------------------------------------------------------------
template <bool USE_TDM>
__global__ __launch_bounds__(512) void csr_fused_kernel(
    const float*    __restrict__ x,
    const _Float16* __restrict__ xh,
    const _Float16* __restrict__ Wt,
    const float*    __restrict__ bias,
    const float*    __restrict__ gamma,
    const float*    __restrict__ beta,
    float*          __restrict__ out) {

    __shared__ __align__(16) char aRaw[25600];   // fits both layouts
    __shared__ float redSum[16], redSq[16], redCnt[16], redNrm[16];
    __shared__ int   redMax[16];

    const int tid  = threadIdx.x;
    const int lane = tid & 31;
    const int wv   = tid >> 5;
    const int half = lane >> 4;
    const int l16  = lane & 15;
    const int m0   = blockIdx.x * 16;
    const int cbase = wv * 256;

    if (tid < 16) {
        redSum[tid] = 0.f; redSq[tid] = 0.f; redCnt[tid] = 0.f;
        redNrm[tid] = 0.f; redMax[tid] = 0;
    }

    // ---- stage A tile (16 rows x 768 f16) into LDS ------------------------
    if constexpr (USE_TDM) {
#if __has_builtin(__builtin_amdgcn_tensor_load_to_lds)
        if (wv == 0) {
            const unsigned long long ga =
                (unsigned long long)(uintptr_t)(xh + (size_t)m0 * D_IN);
            const unsigned int lds = (unsigned int)(uintptr_t)&aRaw[0];
            // group0: count=1 | lds_addr | global_addr | type=2 ("image")
            v4u g0 = {1u, lds, (unsigned int)ga,
                      (unsigned int)((ga >> 32) & 0x01FFFFFFull) | 0x80000000u};
            // group1: data_size=1(2B), pad_enable, pad_interval=7(256dw),
            //         pad_amount=3(4dw); tensor_dim0=768; tensor_dim1=32768;
            //         tile_dim0=768; tile_dim1=16; tensor_dim0_stride=768
            v8i g1 = {(int)0x07D10000u, (int)0x03000000u, (int)0x80000000u,
                      (int)0x03000000u, 16, 768, 0, 0};
            v4i z4 = {0, 0, 0, 0};
#if __clang_major__ >= 23
            v8i z8 = {0, 0, 0, 0, 0, 0, 0, 0};
            __builtin_amdgcn_tensor_load_to_lds(g0, g1, z4, z4, z8, 0);
#else
            __builtin_amdgcn_tensor_load_to_lds(g0, g1, z4, z4, 0);
#endif
#if __has_builtin(__builtin_amdgcn_s_wait_tensorcnt)
            __builtin_amdgcn_s_wait_tensorcnt(0);
#endif
        }
#else
        // toolchain without the TDM builtin: manual staging, same layout
        for (int i = tid; i < 16 * D_IN; i += 512)
            *(_Float16*)(aRaw + tdm_off(i)) = xh[(size_t)m0 * D_IN + i];
#endif
    } else {
        _Float16* aLds = (_Float16*)aRaw;
        for (int i = tid; i < 16 * D_IN; i += 512) {
            const int r = i / D_IN, c = i - r * D_IN;
            aLds[r * A_PITCH + c] = (_Float16)x[(size_t)(m0 + r) * D_IN + c];
        }
    }
    __syncthreads();

    // ---- GEMM: 24 K-steps of 32, 16 tiles per wave, 2-tile-deep pipeline --
    v8f acc[16];
#pragma unroll
    for (int t = 0; t < 16; ++t)
#pragma unroll
        for (int j = 0; j < 8; ++j) acc[t][j] = 0.0f;

    const _Float16* __restrict__ arow = (const _Float16*)aRaw + l16 * A_PITCH;
    const int ibase = l16 * D_IN + half * 16;  // TDM layout linear index base
    const _Float16* __restrict__ brow =
        Wt + (size_t)(cbase + l16) * D_IN + half * 16;

    auto loadA0 = [&](int k) -> v8h {
        if constexpr (USE_TDM) {
            const int i0 = ibase + k;
            return *(const v8h*)(aRaw + tdm_off(i0));
        } else {
            return *(const v8h*)(arow + k + half * 16);
        }
    };
    auto loadA1 = [&](int k) -> v8h {
        if constexpr (USE_TDM) {
            const int i1 = ibase + k + 8;
            return *(const v8h*)(aRaw + tdm_off(i1));
        } else {
            return *(const v8h*)(arow + k + half * 16 + 8);
        }
    };
    auto loadB = [&](int t, int k) -> v16h {
        return *(const v16h*)(brow + (size_t)t * 16 * D_IN + k);
    };

    v16h av;
    {
        v8h a0 = loadA0(0), a1 = loadA1(0);
#pragma unroll
        for (int i = 0; i < 8; ++i) { av[i] = a0[i]; av[i + 8] = a1[i]; }
    }
    v16h b0 = loadB(0, 0);
    v16h b1 = loadB(1, 0);

    for (int kk = 0; kk < D_IN; kk += 32) {
        const int kn = (kk + 32 < D_IN) ? kk + 32 : 0;   // next K (clamped)
        v8h a0n, a1n;
#pragma unroll
        for (int t = 0; t < 16; ++t) {
            if (t == 8) { a0n = loadA0(kn); a1n = loadA1(kn); }
            v16h bn = (t < 14) ? loadB(t + 2, kk) : loadB(t - 14, kn);
            acc[t] = __builtin_amdgcn_wmma_f32_16x16x32_f16(
                false, av, false, b0, (short)0, acc[t], false, false);
            b0 = b1; b1 = bn;
        }
#pragma unroll
        for (int i = 0; i < 8; ++i) { av[i] = a0n[i]; av[i + 8] = a1n[i]; }
    }

    // ---- per-lane column constants ---------------------------------------
    float bia[16], gam[16], bet[16];
#pragma unroll
    for (int t = 0; t < 16; ++t) {
        const int c = cbase + t * 16 + l16;
        bia[t] = bias[c]; gam[t] = gamma[c]; bet[t] = beta[c];
    }

    // ---- bias + LN statistics (rows r = 8*half + j) ----------------------
    float psum[8], psq[8];
#pragma unroll
    for (int j = 0; j < 8; ++j) { psum[j] = 0.f; psq[j] = 0.f; }
#pragma unroll
    for (int t = 0; t < 16; ++t)
#pragma unroll
        for (int j = 0; j < 8; ++j) {
            float v = acc[t][j] + bia[t];
            acc[t][j] = v;
            psum[j] += v; psq[j] += v * v;
        }
#pragma unroll
    for (int m = 1; m < 16; m <<= 1)
#pragma unroll
        for (int j = 0; j < 8; ++j) {
            psum[j] += __shfl_xor(psum[j], m, 32);
            psq[j]  += __shfl_xor(psq[j],  m, 32);
        }
    if (l16 == 0) {
#pragma unroll
        for (int j = 0; j < 8; ++j) {
            atomicAdd(&redSum[half * 8 + j], psum[j]);
            atomicAdd(&redSq [half * 8 + j], psq[j]);
        }
    }
    __syncthreads();

    float mu[8], rstd[8];
#pragma unroll
    for (int j = 0; j < 8; ++j) {
        const int r = half * 8 + j;
        const float m  = redSum[r] * (1.0f / D_OUT);
        const float vv = redSq[r] * (1.0f / D_OUT) - m * m;
        mu[j] = m; rstd[j] = rsqrtf(vv + 1e-5f);
    }

    // ---- normalize in registers; per-row max |p| -------------------------
    float amax[8];
#pragma unroll
    for (int j = 0; j < 8; ++j) amax[j] = 0.f;
#pragma unroll
    for (int t = 0; t < 16; ++t)
#pragma unroll
        for (int j = 0; j < 8; ++j) {
            float p = (acc[t][j] - mu[j]) * rstd[j] * gam[t] + bet[t];
            acc[t][j] = p;
            amax[j] = fmaxf(amax[j], fabsf(p));
        }
#pragma unroll
    for (int m = 1; m < 16; m <<= 1)
#pragma unroll
        for (int j = 0; j < 8; ++j)
            amax[j] = fmaxf(amax[j], __shfl_xor(amax[j], m, 32));
    if (l16 == 0) {
#pragma unroll
        for (int j = 0; j < 8; ++j)
            atomicMax(&redMax[half * 8 + j], __float_as_int(amax[j]));
    }
    __syncthreads();

    // ---- top-64 threshold per row via bisection on count(|p| > T) -------
    float lo[8], hi[8];
#pragma unroll
    for (int j = 0; j < 8; ++j) {
        lo[j] = 0.f;
        hi[j] = __int_as_float(redMax[half * 8 + j]);
    }
    for (int it = 0; it < BISECT_ITERS; ++it) {
        float fc[8];
#pragma unroll
        for (int j = 0; j < 8; ++j) {
            const float mid = 0.5f * (lo[j] + hi[j]);
            float c = 0.f;
#pragma unroll
            for (int t = 0; t < 16; ++t)
                c += (fabsf(acc[t][j]) > mid) ? 1.0f : 0.0f;
            fc[j] = c;
        }
#pragma unroll
        for (int m = 1; m < 16; m <<= 1)
#pragma unroll
            for (int j = 0; j < 8; ++j)
                fc[j] += __shfl_xor(fc[j], m, 32);
        if (l16 == 0) {
#pragma unroll
            for (int j = 0; j < 8; ++j)
                atomicAdd(&redCnt[half * 8 + j], fc[j]);
        }
        __syncthreads();
#pragma unroll
        for (int j = 0; j < 8; ++j) {
            const float mid = 0.5f * (lo[j] + hi[j]);
            if (redCnt[half * 8 + j] >= (float)ACTIVE - 0.5f) lo[j] = mid;
            else hi[j] = mid;
        }
        __syncthreads();
        if (tid < 16) redCnt[tid] = 0.f;
        __syncthreads();
    }

    // ---- L2 norm of kept entries -----------------------------------------
    float ksq[8];
#pragma unroll
    for (int j = 0; j < 8; ++j) ksq[j] = 0.f;
#pragma unroll
    for (int t = 0; t < 16; ++t)
#pragma unroll
        for (int j = 0; j < 8; ++j) {
            const float p = acc[t][j];
            if (fabsf(p) > lo[j]) ksq[j] += p * p;
        }
#pragma unroll
    for (int m = 1; m < 16; m <<= 1)
#pragma unroll
        for (int j = 0; j < 8; ++j)
            ksq[j] += __shfl_xor(ksq[j], m, 32);
    if (l16 == 0) {
#pragma unroll
        for (int j = 0; j < 8; ++j)
            atomicAdd(&redNrm[half * 8 + j], ksq[j]);
    }
    __syncthreads();

    float rinv[8];
#pragma unroll
    for (int j = 0; j < 8; ++j)
        rinv[j] = 1.0f / fmaxf(sqrtf(redNrm[half * 8 + j]), 1e-12f);

    // ---- write sparse-normalized output (all elements) -------------------
#pragma unroll
    for (int t = 0; t < 16; ++t) {
        const int c = cbase + t * 16 + l16;
#pragma unroll
        for (int j = 0; j < 8; ++j) {
            const int r = half * 8 + j;
            const float p = acc[t][j];
            out[(size_t)(m0 + r) * D_OUT + c] =
                (fabsf(p) > lo[j]) ? p * rinv[j] : 0.0f;
        }
    }
}

// ---------------------------------------------------------------------------
extern "C" void kernel_launch(void* const* d_in, const int* in_sizes, int n_in,
                              void* d_out, int out_size, void* d_ws, size_t ws_size,
                              hipStream_t stream) {
    (void)in_sizes; (void)n_in; (void)out_size;
    const float* x     = (const float*)d_in[0];
    const float* W     = (const float*)d_in[1];
    const float* bias  = (const float*)d_in[2];
    const float* gamma = (const float*)d_in[3];
    const float* beta  = (const float*)d_in[4];
    float*       out   = (float*)d_out;

    const size_t wtBytes = (size_t)D_OUT * D_IN * sizeof(_Float16);   // 6 MB
    const size_t xhBytes = (size_t)B_ROWS * D_IN * sizeof(_Float16);  // 48 MB
    _Float16* Wt = (_Float16*)d_ws;

    wconv_kernel<<<dim3(D_OUT / 32, D_IN / 32), dim3(32, 8), 0, stream>>>(W, Wt);

    if (ws_size >= wtBytes + xhBytes) {
        _Float16* xh = (_Float16*)((char*)d_ws + wtBytes);
        xconv_kernel<<<(B_ROWS * D_IN) / (256 * 4), 256, 0, stream>>>(x, xh);
        csr_fused_kernel<true><<<B_ROWS / 16, 512, 0, stream>>>(
            x, xh, Wt, bias, gamma, beta, out);
    } else {
        csr_fused_kernel<false><<<B_ROWS / 16, 512, 0, stream>>>(
            x, nullptr, Wt, bias, gamma, beta, out);
    }
}
